// BertEncoder_49847390437886
// MI455X (gfx1250) — matrix operand order repaired and checked by reading, not compiled, not run on hardware
//
#include <hip/hip_runtime.h>
#include <math.h>

// ---------------------------------------------------------------------------
// Early-exit BERT (switch routing, eval mode) for gfx1250 / MI455X.
// Dense compute of all branches (exactly like the JAX reference), masked
// select per sample. All GEMMs via v_wmma_f32_16x16x32_bf16 (wave32),
// double-buffered LDS staging with single-op packed bf16 conversion.
// ---------------------------------------------------------------------------

#define B_   8
#define S_   256
#define H_   768
#define FF_  3072
#define NH_  12
#define HD_  64
#define NPART_ 6
#define TOK_ (B_ * S_)          // 2048 tokens

typedef __attribute__((ext_vector_type(16))) __bf16 v16bf;
typedef __attribute__((ext_vector_type(8)))  float  v8f;
typedef __attribute__((ext_vector_type(4)))  float  v4f;

// pack two floats into one dword of bf16 (lo, hi).
// Preference: v_cvt_pk_bf16_f32 (RNE) > v_perm_b32 (truncate) > manual.
__device__ __forceinline__ unsigned pkbf(float lo, float hi) {
#if defined(__has_builtin) && __has_builtin(__builtin_amdgcn_cvt_pk_bf16_f32)
  typedef __attribute__((ext_vector_type(2))) __bf16 v2bf;
  union { v2bf v; unsigned u; } u;
  u.v = __builtin_amdgcn_cvt_pk_bf16_f32(lo, hi);
  return u.u;
#else
  union { float f; unsigned u; } a, b;
  a.f = hi; b.f = lo;
#if defined(__has_builtin) && __has_builtin(__builtin_amdgcn_perm)
  // dst = { a[31:16], b[31:16] }  (one v_perm_b32)
  return __builtin_amdgcn_perm(a.u, b.u, 0x07060302u);
#else
  return (b.u >> 16) | (a.u & 0xFFFF0000u);
#endif
#endif
}

union FragAB { v16bf v; v4f q[2]; };

// register staging for one tile (per thread: 16 A floats + 16 B floats)
struct StageRegs { v4f a[4]; v4f b[4]; };

// ---------------------------------------------------------------------------
// Batched/strided GEMM: C[M,N] = A[M,K] * B + bias(optional, over N)
//   b_is_nk == 0: B stored [K][N] row-major (ldb = N-stride)   (x @ W)
//   b_is_nk == 1: B stored [N][K] row-major (ldb = K-stride)   (q @ k^T)
// Two-level batch: gridDim.z = nb1*nb2, z -> (i1 = z/nb2, i2 = z%nb2).
// Block: 256 threads = 8 waves; tile 128x128; K-step 32; double-buffered LDS.
// Wave w: wm = w&1 (M offset *64), wn = w>>1 (N offset *32) -> 4x2 subtiles.
// ---------------------------------------------------------------------------
__device__ __forceinline__ void stage_gload(
    const float* __restrict__ A, const float* __restrict__ Bm,
    int M, int N, int lda, int ldb, int b_is_nk,
    int by, int bx, int k0, int tid, StageRegs& r)
{
  {
    const int row = tid >> 1;
    const int cg  = (tid & 1) * 16;
    int gr = by + row; if (gr > M - 1) gr = M - 1;
    const float* ap = A + (long long)gr * lda + k0 + cg;
    #pragma unroll
    for (int v = 0; v < 4; v++) r.a[v] = *reinterpret_cast<const v4f*>(ap + 4 * v);
  }
  if (b_is_nk) {
    const int row = tid >> 1;
    const int cg  = (tid & 1) * 16;
    const int gn  = bx + row;
    if (gn < N) {
      const float* bp = Bm + (long long)gn * ldb + k0 + cg;
      #pragma unroll
      for (int v = 0; v < 4; v++) r.b[v] = *reinterpret_cast<const v4f*>(bp + 4 * v);
    } else {
      #pragma unroll
      for (int v = 0; v < 4; v++) r.b[v] = v4f{0.f, 0.f, 0.f, 0.f};
    }
  } else {
    // two adjacent K rows, 8 columns each -> enables packed (k,k+1) dword stores
    const int kp   = (tid >> 4) * 2;     // 0,2,..,30
    const int nseg = (tid & 15) * 8;     // 0..120
    const float* b0 = Bm + (long long)(k0 + kp) * ldb + bx + nseg;
    const float* b1 = b0 + ldb;
    if (bx + 128 <= N) {
      r.b[0] = *reinterpret_cast<const v4f*>(b0);
      r.b[1] = *reinterpret_cast<const v4f*>(b0 + 4);
      r.b[2] = *reinterpret_cast<const v4f*>(b1);
      r.b[3] = *reinterpret_cast<const v4f*>(b1 + 4);
    } else {
      #pragma unroll
      for (int j = 0; j < 8; j++) {
        const int gn = bx + nseg + j;
        const float v0 = (gn < N) ? b0[j] : 0.f;
        const float v1 = (gn < N) ? b1[j] : 0.f;
        r.b[j >> 2][j & 3]       = v0;
        r.b[2 + (j >> 2)][j & 3] = v1;
      }
    }
  }
}

__device__ __forceinline__ void stage_store(
    unsigned* __restrict__ lA, unsigned* __restrict__ lB,   // dword views
    int b_is_nk, int tid, const StageRegs& r)
{
  {
    const int row = tid >> 1;
    const int cg8 = (tid & 1) * 8;     // dword offset
    unsigned* d = lA + row * 16 + cg8; // A row stride: 16 dwords (32 bf16)
    #pragma unroll
    for (int v = 0; v < 4; v++) {
      d[v * 2 + 0] = pkbf(r.a[v][0], r.a[v][1]);
      d[v * 2 + 1] = pkbf(r.a[v][2], r.a[v][3]);
    }
  }
  if (b_is_nk) {
    const int row = tid >> 1;
    const int cg8 = (tid & 1) * 8;
    unsigned* d = lB + row * 20 + cg8; // B row stride: 20 dwords (40 bf16)
    #pragma unroll
    for (int v = 0; v < 4; v++) {
      d[v * 2 + 0] = pkbf(r.b[v][0], r.b[v][1]);
      d[v * 2 + 1] = pkbf(r.b[v][2], r.b[v][3]);
    }
  } else {
    const int kp   = (tid >> 4) * 2;
    const int nseg = (tid & 15) * 8;
    #pragma unroll
    for (int j = 0; j < 8; j++) {
      const float v0 = r.b[j >> 2][j & 3];        // row kp
      const float v1 = r.b[2 + (j >> 2)][j & 3];  // row kp+1
      lB[(nseg + j) * 20 + (kp >> 1)] = pkbf(v0, v1);
    }
  }
}

__global__ __launch_bounds__(256) void gemm_bf16_wmma(
    const float* __restrict__ A, const float* __restrict__ Bm,
    const float* __restrict__ bias, float* __restrict__ C,
    int M, int N, int K, int lda, int ldb, int ldc, int b_is_nk, int nb2,
    long long sA1, long long sB1, long long sC1, long long sBias1,
    long long sA2, long long sB2, long long sC2, long long sBias2)
{
  __shared__ unsigned ldsA32[2][128 * 16];   // [buf][m][k/2]  (bf16 pairs)
  __shared__ unsigned ldsB32[2][128 * 20];   // [buf][n][k/2], row stride 20 dwords

  const int z  = blockIdx.z;
  const int i1 = z / nb2, i2 = z % nb2;
  A   += i1 * sA1 + i2 * sA2;
  Bm  += i1 * sB1 + i2 * sB2;
  C   += i1 * sC1 + i2 * sC2;
  const long long bofs = i1 * sBias1 + i2 * sBias2;

  const int tid  = threadIdx.x;
  const int lane = tid & 31;
  const int wave = tid >> 5;
  const int wm   = wave & 1;
  const int wn   = wave >> 1;
  const int bx   = blockIdx.x * 128;
  const int by   = blockIdx.y * 128;
  const int hi   = (lane >> 4) & 1;   // K-half select per ISA 16-bit layouts
  const int l15  = lane & 15;

  v8f acc[4][2];
  #pragma unroll
  for (int mi = 0; mi < 4; mi++)
    #pragma unroll
    for (int ni = 0; ni < 2; ni++)
      #pragma unroll
      for (int e = 0; e < 8; e++) acc[mi][ni][e] = 0.f;

  const int nk = K >> 5;               // K is always a multiple of 32 here
  StageRegs rg;
  stage_gload(A, Bm, M, N, lda, ldb, b_is_nk, by, bx, 0, tid, rg);
  stage_store(ldsA32[0], ldsB32[0], b_is_nk, tid, rg);

  for (int it = 0; it < nk; it++) {
    __syncthreads();                   // stores of tile it visible; reads of it-1 done
    const int buf  = it & 1;
    const bool more = (it + 1) < nk;
    if (more)
      stage_gload(A, Bm, M, N, lda, ldb, b_is_nk, by, bx, (it + 1) * 32, tid, rg);

    const unsigned* lA = ldsA32[buf];
    const unsigned* lB = ldsB32[buf];
    FragAB fa[4], fb[2];
    #pragma unroll
    for (int mi = 0; mi < 4; mi++) {
      const int r = wm * 64 + mi * 16 + l15;
      fa[mi].q[0] = *reinterpret_cast<const v4f*>(lA + r * 16 + hi * 4);      // K 0-7 / 8-15
      fa[mi].q[1] = *reinterpret_cast<const v4f*>(lA + r * 16 + 8 + hi * 4);  // K 16-23 / 24-31
    }
    #pragma unroll
    for (int ni = 0; ni < 2; ni++) {
      const int n = wn * 32 + ni * 16 + l15;
      fb[ni].q[0] = *reinterpret_cast<const v4f*>(lB + n * 20 + hi * 4);      // K 0-7 / 8-15
      fb[ni].q[1] = *reinterpret_cast<const v4f*>(lB + n * 20 + 8 + hi * 4);  // K 16-23 / 24-31
    }
    #pragma unroll
    for (int mi = 0; mi < 4; mi++)
      #pragma unroll
      for (int ni = 0; ni < 2; ni++)
        acc[mi][ni] = __builtin_amdgcn_wmma_f32_16x16x32_bf16(
            false, fa[mi].v, false, fb[ni].v, (short)0, acc[mi][ni], false, false);

    if (more)
      stage_store(ldsA32[buf ^ 1], ldsB32[buf ^ 1], b_is_nk, tid, rg);
  }

  // ---- store C (f32 16x16 C/D layout: VGPR r -> M=r + hi*8, N = lane&15) ----
  #pragma unroll
  for (int mi = 0; mi < 4; mi++) {
    #pragma unroll
    for (int ni = 0; ni < 2; ni++) {
      const int ccol = bx + wn * 32 + ni * 16 + l15;
      if (ccol >= N) continue;
      const float bv = bias ? bias[bofs + ccol] : 0.f;
      const int rbase = by + wm * 64 + mi * 16 + hi * 8;
      #pragma unroll
      for (int r = 0; r < 8; r++) {
        const int crow = rbase + r;
        if (crow < M) C[(long long)crow * ldc + ccol] = acc[mi][ni][r] + bv;
      }
    }
  }
}

// ---------------------------------------------------------------------------
// out = LayerNorm(x + z) * g + b     (one 256-thread block per token, H=768)
// ---------------------------------------------------------------------------
__global__ __launch_bounds__(256) void add_ln_kernel(
    const float* __restrict__ x, const float* __restrict__ z,
    const float* __restrict__ g, const float* __restrict__ b,
    float* __restrict__ out)
{
  __shared__ float red[256];
  const int row = blockIdx.x, tid = threadIdx.x;
  const float* xr = x + (long long)row * H_;
  const float* zr = z + (long long)row * H_;
  float v0[3], s = 0.f;
  #pragma unroll
  for (int e = 0; e < 3; e++) { int c = tid + e * 256; float t = xr[c] + zr[c]; v0[e] = t; s += t; }
  red[tid] = s; __syncthreads();
  for (int o = 128; o > 0; o >>= 1) { if (tid < o) red[tid] += red[tid + o]; __syncthreads(); }
  const float mean = red[0] / (float)H_; __syncthreads();
  float sq = 0.f;
  #pragma unroll
  for (int e = 0; e < 3; e++) { float d = v0[e] - mean; sq += d * d; }
  red[tid] = sq; __syncthreads();
  for (int o = 128; o > 0; o >>= 1) { if (tid < o) red[tid] += red[tid + o]; __syncthreads(); }
  const float rs = rsqrtf(red[0] / (float)H_ + 1e-12f);
  float* orow = out + (long long)row * H_;
  #pragma unroll
  for (int e = 0; e < 3; e++) { int c = tid + e * 256; orow[c] = (v0[e] - mean) * rs * g[c] + b[c]; }
}

// softmax over last dim with scale + additive mask; one block per score row
__global__ __launch_bounds__(256) void softmax_kernel(
    float* __restrict__ sc, const float* __restrict__ mask, float scale)
{
  __shared__ float red[256];
  const int row = blockIdx.x;             // ((b*NH + h)*S + i)
  const int tid = threadIdx.x;            // j
  const int b = row / (NH_ * S_);
  float* r = sc + (long long)row * S_;
  float v = r[tid] * scale + mask[b * S_ + tid];
  red[tid] = v; __syncthreads();
  for (int o = 128; o > 0; o >>= 1) { if (tid < o) red[tid] = fmaxf(red[tid], red[tid + o]); __syncthreads(); }
  const float mx = red[0]; __syncthreads();
  const float e = __expf(v - mx);
  red[tid] = e; __syncthreads();
  for (int o = 128; o > 0; o >>= 1) { if (tid < o) red[tid] += red[tid + o]; __syncthreads(); }
  r[tid] = e / red[0];
}

__global__ void gelu_kernel(float* __restrict__ x, long long n) {
  long long i = (long long)blockIdx.x * 256 + threadIdx.x;
  if (i < n) { float v = x[i]; x[i] = 0.5f * v * (1.f + erff(v * 0.70710678118654752f)); }
}

// pooled[b*H + n] = tanh( h[b,0,:] @ pw[:,n] + pb[n] )
__global__ void pool_tanh_kernel(const float* __restrict__ h, const float* __restrict__ pw,
                                 const float* __restrict__ pb, float* __restrict__ out) {
  const int idx = blockIdx.x * 256 + threadIdx.x;
  if (idx >= B_ * H_) return;
  const int b = idx / H_, n = idx % H_;
  const float* hr = h + (long long)b * S_ * H_;
  float s = pb[n];
  for (int k = 0; k < H_; k++) s += hr[k] * pw[k * H_ + n];
  out[idx] = tanhf(s);
}

// one wave per sample: router logits/softmax/argmax + exit head bookkeeping
__global__ void router_kernel(
    const float* __restrict__ pooledA, const float* __restrict__ acw, const float* __restrict__ acb, int na,
    const float* __restrict__ pooledE, const float* __restrict__ ecw, const float* __restrict__ ecb, int ne,
    int* __restrict__ active, int* __restrict__ action,
    float* __restrict__ out_probs, float* __restrict__ out_acts,
    float* __restrict__ out_exit_logits, float* __restrict__ out_exit_part, int part)
{
  const int b = blockIdx.x, lane = threadIdx.x;
  float la[4], le[4];
  for (int n = 0; n < na; n++) {
    float s = 0.f;
    for (int k = lane; k < H_; k += 32) s += pooledA[b * H_ + k] * acw[k * na + n];
    for (int o = 16; o > 0; o >>= 1) s += __shfl_xor(s, o, 32);
    la[n] = s + acb[n];
  }
  for (int n = 0; n < ne; n++) {
    float s = 0.f;
    for (int k = lane; k < H_; k += 32) s += pooledE[b * H_ + k] * ecw[k * ne + n];
    for (int o = 16; o > 0; o >>= 1) s += __shfl_xor(s, o, 32);
    le[n] = s + ecb[n];
  }
  if (lane == 0) {
    float mx = la[0]; int act = 0;
    for (int n = 1; n < na; n++) if (la[n] > mx) { mx = la[n]; act = n; }
    float den = 0.f;
    for (int n = 0; n < na; n++) den += __expf(la[n] - mx);
    const int isact = active[b];
    for (int n = 0; n < na; n++)
      out_probs[part * B_ * na + b * na + n] = isact ? (__expf(la[n] - mx) / den) : 1.f;
    out_acts[part * B_ + b] = isact ? (float)act : 0.f;
    if (isact && act == 0) {
      for (int n = 0; n < ne; n++) out_exit_logits[b * ne + n] = le[n];
      out_exit_part[b] = (float)part;
    }
    action[b] = act;
  }
}

__global__ void select_kernel(float* __restrict__ h, const float* __restrict__ base,
                              const float* __restrict__ large,
                              const int* __restrict__ active, const int* __restrict__ action) {
  long long i = (long long)blockIdx.x * 256 + threadIdx.x;
  if (i >= (long long)TOK_ * H_) return;
  const int s = (int)(i / ((long long)S_ * H_));
  if (active[s]) {
    const int a = action[s];
    if (a == 1) h[i] = base[i];
    else if (a == 2) h[i] = large[i];
  }
}

__global__ void update_active_kernel(int* active, const int* action) {
  const int t = threadIdx.x;
  if (t < B_ && action[t] == 0) active[t] = 0;
}

__global__ void init_kernel(int* active, float* exit_logits, float* exit_part) {
  const int t = threadIdx.x;
  if (t < B_) { active[t] = 1; exit_part[t] = -1.f; }
  if (t < B_ * 2) exit_logits[t] = 0.f;
}

__global__ void write_active_kernel(const int* active, float* out) {
  const int t = threadIdx.x;
  if (t < B_) out[t] = (float)active[t];
}

// ---------------------------------------------------------------------------
// Host-side full BERT layer: in -> out  (out may alias in)
// ---------------------------------------------------------------------------
static void run_layer(const float* in, float* out,
                      const float* aw, const float* ab, const float* lng, const float* lnb,
                      const float* wi, const float* bi, const float* wo, const float* bo,
                      float* Q, float* Kb, float* V, float* SC, float* CTX,
                      float* T1, float* T2, float* FFH,
                      const float* mask, hipStream_t stream)
{
  const dim3 blk(256);
  // q,k,v = x @ aw[0..2] + ab[0..2]   (batch=3; Q,K,V contiguous in ws)
  gemm_bf16_wmma<<<dim3(H_ / 128, TOK_ / 128, 3), blk, 0, stream>>>(
      in, aw, ab, Q, TOK_, H_, H_, H_, H_, H_, 0, 1,
      0LL, (long long)H_ * H_, (long long)TOK_ * H_, (long long)H_, 0LL, 0LL, 0LL, 0LL);
  // scores = q @ k^T   (96 batches of 256x256x64; B given as [N=j][K=d])
  gemm_bf16_wmma<<<dim3(2, 2, B_ * NH_), blk, 0, stream>>>(
      Q, Kb, nullptr, SC, S_, S_, HD_, H_, H_, S_, 1, NH_,
      (long long)S_ * H_, (long long)S_ * H_, (long long)NH_ * S_ * S_, 0LL,
      (long long)HD_, (long long)HD_, (long long)S_ * S_, 0LL);
  softmax_kernel<<<B_ * NH_ * S_, 256, 0, stream>>>(SC, mask, 0.125f);
  // ctx = p @ v        (96 batches of 256x64x256)
  gemm_bf16_wmma<<<dim3(1, 2, B_ * NH_), blk, 0, stream>>>(
      SC, V, nullptr, CTX, S_, HD_, S_, S_, H_, H_, 0, NH_,
      (long long)NH_ * S_ * S_, (long long)S_ * H_, (long long)S_ * H_, 0LL,
      (long long)S_ * S_, (long long)HD_, (long long)HD_, 0LL);
  // attn_out = ctx @ aw[3] + ab[3]
  gemm_bf16_wmma<<<dim3(H_ / 128, TOK_ / 128, 1), blk, 0, stream>>>(
      CTX, aw + 3LL * H_ * H_, ab + 3 * H_, T1, TOK_, H_, H_, H_, H_, H_, 0, 1,
      0LL, 0LL, 0LL, 0LL, 0LL, 0LL, 0LL, 0LL);
  add_ln_kernel<<<TOK_, 256, 0, stream>>>(in, T1, lng, lnb, T2);
  // ffn
  gemm_bf16_wmma<<<dim3(FF_ / 128, TOK_ / 128, 1), blk, 0, stream>>>(
      T2, wi, bi, FFH, TOK_, FF_, H_, H_, FF_, FF_, 0, 1,
      0LL, 0LL, 0LL, 0LL, 0LL, 0LL, 0LL, 0LL);
  gelu_kernel<<<(int)(((long long)TOK_ * FF_ + 255) / 256), 256, 0, stream>>>(FFH, (long long)TOK_ * FF_);
  gemm_bf16_wmma<<<dim3(H_ / 128, TOK_ / 128, 1), blk, 0, stream>>>(
      FFH, wo, bo, T1, TOK_, H_, FF_, FF_, H_, H_, 0, 1,
      0LL, 0LL, 0LL, 0LL, 0LL, 0LL, 0LL, 0LL);
  add_ln_kernel<<<TOK_, 256, 0, stream>>>(T2, T1, lng + H_, lnb + H_, out);
}

extern "C" void kernel_launch(void* const* d_in, const int* in_sizes, int n_in,
                              void* d_out, int out_size, void* d_ws, size_t ws_size,
                              hipStream_t stream) {
  (void)in_sizes; (void)n_in; (void)out_size; (void)ws_size;
  const float* hidden = (const float*)d_in[0];
  const float* mask   = (const float*)d_in[1];
  const float* L_aw   = (const float*)d_in[2];
  const float* L_ab   = (const float*)d_in[3];
  const float* L_lng  = (const float*)d_in[4];
  const float* L_lnb  = (const float*)d_in[5];
  const float* L_wi   = (const float*)d_in[6];
  const float* L_bi   = (const float*)d_in[7];
  const float* L_wo   = (const float*)d_in[8];
  const float* L_bo   = (const float*)d_in[9];
  const float* S_aw   = (const float*)d_in[10];
  const float* S_ab   = (const float*)d_in[11];
  const float* S_lng  = (const float*)d_in[12];
  const float* S_lnb  = (const float*)d_in[13];
  const float* S_wi   = (const float*)d_in[14];
  const float* S_bi   = (const float*)d_in[15];
  const float* S_wo   = (const float*)d_in[16];
  const float* S_bo   = (const float*)d_in[17];
  const float* E_pw   = (const float*)d_in[18];
  const float* E_pb   = (const float*)d_in[19];
  const float* E_cw   = (const float*)d_in[20];
  const float* E_cb   = (const float*)d_in[21];
  const float* A_pw   = (const float*)d_in[22];
  const float* A_pb   = (const float*)d_in[23];
  const float* A_cw   = (const float*)d_in[24];
  const float* A_cb   = (const float*)d_in[25];

  float* ws = (float*)d_ws;
  const long long TH = (long long)TOK_ * H_;           // 1572864
  float* Hc   = ws;                                    // current hidden
  float* BASE = ws + 1 * TH;
  float* LAR  = ws + 2 * TH;
  float* Q    = ws + 3 * TH;
  float* Kb   = ws + 4 * TH;
  float* V    = ws + 5 * TH;
  float* CTX  = ws + 6 * TH;
  float* T1   = ws + 7 * TH;
  float* T2   = ws + 8 * TH;
  float* FFH  = ws + 9 * TH;                           // TOK_*FF_ = 4*TH
  float* SC   = ws + 13 * TH;                          // B*NH*S*S = 4*TH
  float* PA   = ws + 17 * TH;                          // 8*768
  float* PE   = PA + B_ * H_;
  int*   active = (int*)(PE + B_ * H_);
  int*   action = active + B_;

  float* out       = (float*)d_out;
  float* out_h     = out;                              // (8,256,768)
  float* out_act   = out + TH;                         // (8,)
  float* out_probs = out_act + B_;                     // (6,8,3)
  float* out_acts  = out_probs + NPART_ * B_ * 3;      // (6,8)
  float* out_exl   = out_acts + NPART_ * B_;           // (8,2)
  float* out_exp   = out_exl + B_ * 2;                 // (8,)

  hipMemcpyAsync(Hc, hidden, TH * sizeof(float), hipMemcpyDeviceToDevice, stream);
  init_kernel<<<1, 32, 0, stream>>>(active, out_exl, out_exp);

  for (int p = 0; p < NPART_; p++) {
    // router + exit heads (pool CLS -> tanh -> linear)
    pool_tanh_kernel<<<(B_ * H_ + 255) / 256, 256, 0, stream>>>(Hc, A_pw, A_pb, PA);
    pool_tanh_kernel<<<(B_ * H_ + 255) / 256, 256, 0, stream>>>(
        Hc, E_pw + (long long)p * H_ * H_, E_pb + p * H_, PE);
    router_kernel<<<B_, 32, 0, stream>>>(
        PA, A_cw, A_cb, 3, PE, E_cw + (long long)p * H_ * 2, E_cb + p * 2, 2,
        active, action, out_probs, out_acts, out_exl, out_exp, p);

    // small branch: one compressed layer
    run_layer(Hc, BASE,
              S_aw + (long long)p * 4 * H_ * H_, S_ab + (long long)p * 4 * H_,
              S_lng + (long long)p * 2 * H_,     S_lnb + (long long)p * 2 * H_,
              S_wi + (long long)p * H_ * FF_,    S_bi + (long long)p * FF_,
              S_wo + (long long)p * FF_ * H_,    S_bo + (long long)p * H_,
              Q, Kb, V, SC, CTX, T1, T2, FFH, mask, stream);
    // large branch: two full layers
    for (int off = 0; off < 2; off++) {
      const long long j = 2LL * p + off;
      run_layer(off == 0 ? Hc : LAR, LAR,
                L_aw + j * 4 * H_ * H_, L_ab + j * 4 * H_,
                L_lng + j * 2 * H_,     L_lnb + j * 2 * H_,
                L_wi + j * H_ * FF_,    L_bi + j * FF_,
                L_wo + j * FF_ * H_,    L_bo + j * H_,
                Q, Kb, V, SC, CTX, T1, T2, FFH, mask, stream);
    }
    // masked select, then deactivate exiting samples
    select_kernel<<<(int)((TH + 255) / 256), 256, 0, stream>>>(Hc, BASE, LAR, active, action);
    update_active_kernel<<<1, 32, 0, stream>>>(active, action);
  }

  write_active_kernel<<<1, 32, 0, stream>>>(active, out_act);
  hipMemcpyAsync(out_h, Hc, TH * sizeof(float), hipMemcpyDeviceToDevice, stream);
}
